// CausalSelfAttention_6828998000788
// MI455X (gfx1250) — matrix operand order repaired and checked by reading, not compiled, not run on hardware
//
#include <hip/hip_runtime.h>

// ---------------------------------------------------------------------------
// Causal self-attention (RoPE + GQA) for MI455X / gfx1250.
// f16 WMMA (v_wmma_f32_16x16x32_f16) everywhere, f32 accumulate.
// Dense GEMMs: workgroup 64x64 tiles double-buffered in LDS, filled by the
// Tensor Data Mover (tensor_load_to_lds, TENSORcnt-fenced). Attention:
// flash-style online softmax, 1 wave per 16-query tile.
// Fragment loads of a whole K-chunk are issued as one clause ahead of the
// WMMA burst so s_wait_dscnt becomes partial instead of a full drain.
// ---------------------------------------------------------------------------

typedef _Float16 h16;
typedef __attribute__((ext_vector_type(16))) _Float16 v16h;
typedef __attribute__((ext_vector_type(8)))  _Float16 v8h;
typedef __attribute__((ext_vector_type(4)))  _Float16 v4h;
typedef __attribute__((ext_vector_type(8)))  float    v8f;
typedef __attribute__((ext_vector_type(4)))  unsigned int v4u;
typedef __attribute__((ext_vector_type(8)))  int v8i;
typedef __attribute__((ext_vector_type(4)))  int v4i;

#define T_SEQ  2048
#define DMODEL 2048
#define NHEAD  32
#define NKV    8
#define HDIM   64
#define LN_ROPE_BASE 9.210340371976184f   // ln(10000)

// ---------------- WMMA fragment helpers (wave32, ISA 7.12.2 layouts) -------
__device__ __forceinline__ v16h load_a_frag(const h16* A, int ld, int m0,
                                            int k0, int lane) {
  const h16* p = A + (size_t)(m0 + (lane & 15)) * ld + k0 + (lane >> 4) * 8;
  v8h lo = *(const v8h*)p;
  v8h hi = *(const v8h*)(p + 16);
  v16h r;
#pragma unroll
  for (int i = 0; i < 8; ++i) { r[i] = lo[i]; r[i + 8] = hi[i]; }
  return r;
}

__device__ __forceinline__ v16h load_b_frag(const h16* W, int ld, int n0,
                                            int k0, int lane) {
  const h16* p = W + (size_t)(n0 + (lane & 15)) * ld + k0 + (lane >> 4) * 8;
  v8h lo = *(const v8h*)p;
  v8h hi = *(const v8h*)(p + 16);
  v16h r;
#pragma unroll
  for (int i = 0; i < 8; ++i) { r[i] = lo[i]; r[i + 8] = hi[i]; }
  return r;
}

__device__ __forceinline__ v8f wmma_f16(v16h a, v16h b, v8f c) {
  return __builtin_amdgcn_wmma_f32_16x16x32_f16(false, a, false, b,
                                                (short)0, c, false, false);
}

__device__ __forceinline__ float half_row_max(float x) {
#pragma unroll
  for (int m = 1; m <= 8; m <<= 1) x = fmaxf(x, __shfl_xor(x, m, 32));
  return x;
}
__device__ __forceinline__ float half_row_sum(float x) {
#pragma unroll
  for (int m = 1; m <= 8; m <<= 1) x += __shfl_xor(x, m, 32);
  return x;
}

// ---------------- Tensor Data Mover: 2D tile -> LDS ------------------------
__device__ __forceinline__ void tdm_load_2d(const h16* lds_dst,
                                            const h16* gsrc,
                                            unsigned tile_rows,
                                            unsigned tile_cols,
                                            unsigned tensor_d0,
                                            unsigned tensor_d1,
                                            unsigned stride_elems) {
  unsigned lds_off = (unsigned)(unsigned long long)lds_dst;  // addr[31:0]
  unsigned long long ga = (unsigned long long)gsrc;
  v4u g0;
  g0[0] = 1u;                                        // count=1 (valid user D#)
  g0[1] = lds_off;                                   // lds_addr (bytes)
  g0[2] = (unsigned)(ga & 0xFFFFFFFFu);              // global_addr[31:0]
  g0[3] = (unsigned)((ga >> 32) & 0x01FFFFFFu)       // global_addr[56:32]
          | 0x80000000u;                             // type=2 ("image")
  v8i g1;
  g1[0] = (int)(1u << 16);                           // data_size=1 -> 2 bytes
  g1[1] = (int)((tensor_d0 & 0xFFFFu) << 16);        // tensor_dim0[15:0]
  g1[2] = (int)(((tensor_d0 >> 16) & 0xFFFFu)
          | ((tensor_d1 & 0xFFFFu) << 16));          // dim0[31:16] | dim1[15:0]
  g1[3] = (int)(((tensor_d1 >> 16) & 0xFFFFu)
          | ((tile_cols & 0xFFFFu) << 16));          // dim1[31:16] | tile_dim0
  g1[4] = (int)(tile_rows & 0xFFFFu);                // tile_dim1 (tile_dim2=0)
  g1[5] = (int)stride_elems;                         // tensor_dim0_stride[31:0]
  g1[6] = 0;
  g1[7] = 0;
  v4i z = {0, 0, 0, 0};
#if defined(__clang_major__) && __clang_major__ >= 23
  v8i z8 = {0, 0, 0, 0, 0, 0, 0, 0};
  __builtin_amdgcn_tensor_load_to_lds(g0, g1, z, z, z8, 0);
#else
  __builtin_amdgcn_tensor_load_to_lds(g0, g1, z, z, 0);
#endif
}

// ---------------- Kernel 1: fp32 -> f16 conversion -------------------------
__global__ void cvt_kernel(const float* __restrict__ in, h16* __restrict__ out,
                           int n4) {
  int stride = gridDim.x * blockDim.x;
  for (int j = blockIdx.x * blockDim.x + threadIdx.x; j < n4; j += stride) {
    float4 v = ((const float4*)in)[j];
    v4h o = { (h16)v.x, (h16)v.y, (h16)v.z, (h16)v.w };
    ((v4h*)out)[j] = o;
  }
}

// ---------------- shared GEMM block body -----------------------------------
// Loads the full K=64 chunk (2 A frags + 8 B frags) before the 8 WMMAs.
__device__ __forceinline__ void gemm_chunk(const h16* At, const h16* Bt,
                                           int wid, int lane, v8f acc[4]) {
  v16h a0 = load_a_frag(At, 64, wid * 16, 0, lane);
  v16h a1 = load_a_frag(At, 64, wid * 16, 32, lane);
  v16h b[8];
#pragma unroll
  for (int j = 0; j < 4; ++j) {
    b[j]     = load_b_frag(Bt, 64, j * 16, 0, lane);
    b[4 + j] = load_b_frag(Bt, 64, j * 16, 32, lane);
  }
#pragma unroll
  for (int j = 0; j < 4; ++j) acc[j] = wmma_f16(a0, b[j], acc[j]);
#pragma unroll
  for (int j = 0; j < 4; ++j) acc[j] = wmma_f16(a1, b[4 + j], acc[j]);
}

// ---------------- Kernel 2: fused QKV projection + RoPE --------------------
__global__ void __launch_bounds__(128)
qkv_rope_kernel(const h16* __restrict__ xh, const h16* __restrict__ wqh,
                const h16* __restrict__ wkh, const h16* __restrict__ wvh,
                h16* __restrict__ qout, h16* __restrict__ kout,
                h16* __restrict__ vtout) {
  __shared__ h16 ldsA[2][64 * 64];   // x tile (rows t, ld=64)
  __shared__ h16 ldsB[2][64 * 64];   // W tile (rows n, ld=64)
  int wid = threadIdx.x >> 5, lane = threadIdx.x & 31;
  int t0 = blockIdx.y * 64;
  int nblock = blockIdx.x * 64;      // region boundaries are multiples of 64

  const h16* W; int nw0; unsigned wrows;
  if (nblock < DMODEL)              { W = wqh; nw0 = nblock;               wrows = DMODEL; }
  else if (nblock < DMODEL + 512)   { W = wkh; nw0 = nblock - DMODEL;      wrows = 512; }
  else                              { W = wvh; nw0 = nblock - DMODEL - 512; wrows = 512; }

  const int KT = DMODEL / 64;
  if (wid == 0) {
    tdm_load_2d(&ldsA[0][0], xh + (size_t)t0 * DMODEL, 64, 64, DMODEL, T_SEQ, DMODEL);
    tdm_load_2d(&ldsB[0][0], W + (size_t)nw0 * DMODEL, 64, 64, DMODEL, wrows, DMODEL);
  }

  v8f acc[4] = {{}, {}, {}, {}};
  for (int it = 0; it < KT; ++it) {
    if (wid == 0) __builtin_amdgcn_s_wait_tensorcnt(0);
    __syncthreads();  // publish buffer (it&1); reads of ((it+1)&1) are done
    if (wid == 0 && it + 1 < KT) {
      int k0 = (it + 1) * 64, nb = (it + 1) & 1;
      tdm_load_2d(&ldsA[nb][0], xh + (size_t)t0 * DMODEL + k0, 64, 64, DMODEL, T_SEQ, DMODEL);
      tdm_load_2d(&ldsB[nb][0], W + (size_t)nw0 * DMODEL + k0, 64, 64, DMODEL, wrows, DMODEL);
    }
    int cb = it & 1;
    gemm_chunk(&ldsA[cb][0], &ldsB[cb][0], wid, lane, acc);
  }

  // ---- epilogue: RoPE for q/k, transpose-store for v
  int nc = lane & 15;
  int mbase = (lane >> 4) * 8;
  int trow0 = t0 + wid * 16;
#pragma unroll
  for (int j = 0; j < 4; ++j) {
    int ncol = nblock + j * 16 + nc;   // absolute column in [0,3072)
    if (ncol < DMODEL + NKV * HDIM) {
      int n = (ncol < DMODEL) ? ncol : (ncol - DMODEL);
      int hd = n & (HDIM - 1);
      int jj = hd >> 1;
      float theta = __expf(-((float)(2 * jj) / (float)HDIM) * LN_ROPE_BASE);
      float sgn = (hd & 1) ? 1.0f : -1.0f;
      int head = n >> 6;
      h16* dst = (ncol < DMODEL) ? (qout + (size_t)head * T_SEQ * HDIM)
                                 : (kout + (size_t)head * T_SEQ * HDIM);
#pragma unroll
      for (int r = 0; r < 8; ++r) {
        float c = acc[j][r];
        float partner = __shfl_xor(c, 1, 32);  // even<->odd column pair
        int t = trow0 + mbase + r;
        float sn, cs;
        __sincosf((float)t * theta, &sn, &cs);
        dst[(size_t)t * HDIM + hd] = (h16)(c * cs + sgn * partner * sn);
      }
    } else {
      int n = ncol - DMODEL - NKV * HDIM;
      int head = n >> 6, hd = n & 63;
#pragma unroll
      for (int r = 0; r < 8; ++r) {
        int t = trow0 + mbase + r;
        vtout[((size_t)head * HDIM + hd) * T_SEQ + t] = (h16)acc[j][r];
      }
    }
  }
}

// ---------------- Kernel 3: flash attention (causal, GQA) ------------------
__global__ void attn_kernel(const h16* __restrict__ q,
                            const h16* __restrict__ kmat,
                            const h16* __restrict__ vt,
                            h16* __restrict__ aout) {  // [T][D] f16
  __shared__ h16 pbuf[4][16 * 32];  // per-wave P transpose staging
  int lwave = threadIdx.x >> 5;
  int lane  = threadIdx.x & 31;
  int gwave = blockIdx.x * (blockDim.x >> 5) + lwave;
  int h  = gwave / (T_SEQ / 16);
  int mt = gwave % (T_SEQ / 16);
  if (h >= NHEAD) return;
  int t0  = mt * 16;
  int kvh = h / (NHEAD / NKV);
  const h16* qh = q    + (size_t)h   * T_SEQ * HDIM;
  const h16* kh = kmat + (size_t)kvh * T_SEQ * HDIM;
  const h16* vh = vt   + (size_t)kvh * HDIM * T_SEQ;

  v16h aq0 = load_a_frag(qh, HDIM, t0, 0, lane);
  v16h aq1 = load_a_frag(qh, HDIM, t0, 32, lane);
#pragma unroll
  for (int i = 0; i < 16; ++i) {
    aq0[i] = aq0[i] * (h16)0.125f;   // 1/sqrt(64)
    aq1[i] = aq1[i] * (h16)0.125f;
  }

  v8f o0 = {}, o1 = {}, o2 = {}, o3 = {};
  float mrow[8], lrow[8];
#pragma unroll
  for (int r = 0; r < 8; ++r) { mrow[r] = -3.0e38f; lrow[r] = 0.0f; }

  int mbase = (lane >> 4) * 8;
  int nc = lane & 15;
  int nblocks = (t0 + 16 + 31) / 32;
  h16* pb = &pbuf[lwave][0];

  for (int blk = 0; blk < nblocks; ++blk) {
    int s0 = blk * 32;
    // hoist all 4 K-matrix B fragments, then the 4 score WMMAs
    v16h bk0 = load_b_frag(kh, HDIM, s0,      0,  lane);
    v16h bk1 = load_b_frag(kh, HDIM, s0,      32, lane);
    v16h bk2 = load_b_frag(kh, HDIM, s0 + 16, 0,  lane);
    v16h bk3 = load_b_frag(kh, HDIM, s0 + 16, 32, lane);
    v8f s_lo = {}, s_hi = {};
    s_lo = wmma_f16(aq0, bk0, s_lo);
    s_lo = wmma_f16(aq1, bk1, s_lo);
    s_hi = wmma_f16(aq0, bk2, s_hi);
    s_hi = wmma_f16(aq1, bk3, s_hi);

    // V B-fragments can load while softmax VALU runs
    v16h bv0 = load_b_frag(vh + s0, T_SEQ, 0,  0, lane);
    v16h bv1 = load_b_frag(vh + s0, T_SEQ, 16, 0, lane);
    v16h bv2 = load_b_frag(vh + s0, T_SEQ, 32, 0, lane);
    v16h bv3 = load_b_frag(vh + s0, T_SEQ, 48, 0, lane);

#pragma unroll
    for (int r = 0; r < 8; ++r) {
      int trow = t0 + mbase + r;
      float x0 = (s0 + nc      <= trow) ? s_lo[r] : -3.0e38f;
      float x1 = (s0 + 16 + nc <= trow) ? s_hi[r] : -3.0e38f;
      float bm   = half_row_max(fmaxf(x0, x1));
      float mnew = fmaxf(mrow[r], bm);
      float scl  = __expf(mrow[r] - mnew);
      float p0   = __expf(x0 - mnew);
      float p1   = __expf(x1 - mnew);
      lrow[r] = lrow[r] * scl + half_row_sum(p0 + p1);
      mrow[r] = mnew;
      o0[r] *= scl; o1[r] *= scl; o2[r] *= scl; o3[r] *= scl;
      pb[(mbase + r) * 32 + nc]      = (h16)p0;  // C-layout -> row-major P
      pb[(mbase + r) * 32 + nc + 16] = (h16)p1;
    }
    v16h pa = load_a_frag(pb, 32, 0, 0, lane);   // in-wave LDS transpose
    o0 = wmma_f16(pa, bv0, o0);
    o1 = wmma_f16(pa, bv1, o1);
    o2 = wmma_f16(pa, bv2, o2);
    o3 = wmma_f16(pa, bv3, o3);
  }

#pragma unroll
  for (int r = 0; r < 8; ++r) {
    float inv = 1.0f / lrow[r];
    size_t base = (size_t)(t0 + mbase + r) * DMODEL + (size_t)h * HDIM;
    aout[base + 0  + nc] = (h16)(o0[r] * inv);
    aout[base + 16 + nc] = (h16)(o1[r] * inv);
    aout[base + 32 + nc] = (h16)(o2[r] * inv);
    aout[base + 48 + nc] = (h16)(o3[r] * inv);
  }
}

// ---------------- Kernel 4: output projection (TDM-staged GEMM -> f32) -----
__global__ void __launch_bounds__(128)
outproj_kernel(const h16* __restrict__ ah, const h16* __restrict__ woh,
               float* __restrict__ out) {
  __shared__ h16 ldsA[2][64 * 64];
  __shared__ h16 ldsB[2][64 * 64];
  int wid = threadIdx.x >> 5, lane = threadIdx.x & 31;
  int t0 = blockIdx.y * 64;
  int n0 = blockIdx.x * 64;

  const int KT = DMODEL / 64;
  if (wid == 0) {
    tdm_load_2d(&ldsA[0][0], ah + (size_t)t0 * DMODEL, 64, 64, DMODEL, T_SEQ, DMODEL);
    tdm_load_2d(&ldsB[0][0], woh + (size_t)n0 * DMODEL, 64, 64, DMODEL, DMODEL, DMODEL);
  }

  v8f acc[4] = {{}, {}, {}, {}};
  for (int it = 0; it < KT; ++it) {
    if (wid == 0) __builtin_amdgcn_s_wait_tensorcnt(0);
    __syncthreads();
    if (wid == 0 && it + 1 < KT) {
      int k0 = (it + 1) * 64, nb = (it + 1) & 1;
      tdm_load_2d(&ldsA[nb][0], ah + (size_t)t0 * DMODEL + k0, 64, 64, DMODEL, T_SEQ, DMODEL);
      tdm_load_2d(&ldsB[nb][0], woh + (size_t)n0 * DMODEL + k0, 64, 64, DMODEL, DMODEL, DMODEL);
    }
    int cb = it & 1;
    gemm_chunk(&ldsA[cb][0], &ldsB[cb][0], wid, lane, acc);
  }

  int nc = lane & 15, mbase = (lane >> 4) * 8;
#pragma unroll
  for (int j = 0; j < 4; ++j)
#pragma unroll
    for (int r = 0; r < 8; ++r)
      out[(size_t)(t0 + wid * 16 + mbase + r) * DMODEL + n0 + j * 16 + nc] = acc[j][r];
}

// ---------------- host-side launch -----------------------------------------
extern "C" void kernel_launch(void* const* d_in, const int* in_sizes, int n_in,
                              void* d_out, int out_size, void* d_ws,
                              size_t ws_size, hipStream_t stream) {
  (void)in_sizes; (void)n_in; (void)out_size; (void)ws_size;
  const float* x  = (const float*)d_in[0];
  const float* wq = (const float*)d_in[1];
  const float* wk = (const float*)d_in[2];
  const float* wv = (const float*)d_in[3];
  const float* wo = (const float*)d_in[4];
  float* out = (float*)d_out;

  // workspace layout (f16 elements); total ~48 MB (fits 192 MB L2)
  h16* xh   = (h16*)d_ws;
  h16* wqh  = xh   + (size_t)T_SEQ * DMODEL;
  h16* wkh  = wqh  + (size_t)DMODEL * DMODEL;
  h16* wvh  = wkh  + (size_t)NKV * HDIM * DMODEL;
  h16* woh  = wvh  + (size_t)NKV * HDIM * DMODEL;
  h16* qws  = woh  + (size_t)DMODEL * DMODEL;
  h16* kws  = qws  + (size_t)NHEAD * T_SEQ * HDIM;
  h16* vtws = kws  + (size_t)NKV * T_SEQ * HDIM;
  h16* aws  = vtws + (size_t)NKV * T_SEQ * HDIM;

  cvt_kernel<<<1024, 256, 0, stream>>>(x,  xh,  (T_SEQ * DMODEL) / 4);
  cvt_kernel<<<1024, 256, 0, stream>>>(wq, wqh, (DMODEL * DMODEL) / 4);
  cvt_kernel<<<512,  256, 0, stream>>>(wk, wkh, (NKV * HDIM * DMODEL) / 4);
  cvt_kernel<<<512,  256, 0, stream>>>(wv, wvh, (NKV * HDIM * DMODEL) / 4);
  cvt_kernel<<<1024, 256, 0, stream>>>(wo, woh, (DMODEL * DMODEL) / 4);

  // QKV+RoPE: 48 x 32 blocks of 64x64 over [2048 x 3072]
  qkv_rope_kernel<<<dim3(48, 32), 128, 0, stream>>>(xh, wqh, wkh, wvh,
                                                    qws, kws, vtws);

  // flash attention: 32 heads x 128 query tiles, 4 waves/block
  attn_kernel<<<(NHEAD * 128) / 4, 128, 0, stream>>>(qws, kws, vtws, aws);

  // output projection: 32 x 32 blocks of 64x64 over [2048 x 2048]
  outproj_kernel<<<dim3(32, 32), 128, 0, stream>>>(aws, woh, out);
}
